// Encoder_60885456388865
// MI455X (gfx1250) — matrix operand order repaired and checked by reading, not compiled, not run on hardware
//
#include <hip/hip_runtime.h>
#include <hip/hip_bf16.h>

// ---------------------------------------------------------------------------
// 2-layer GRU (TF GRUCell semantics) for MI455X / gfx1250.
//   B=32, S=512, V=32000, E=256, U=1024.
// Strategy:
//   * Split W = [Wx ; Wh]. Precompute Gx = x@Wgx+bg, Cx = x@Wcx+bc with a big
//     parallel bf16 WMMA GEMM (M = S*B = 16384 rows, s-major so each timestep
//     is 2 contiguous 16-row tiles).
//   * Recurrence: per step, phase-1 computes gates gi = Gx[t] + h@Whg,
//     sigmoid -> r,u; stores (r*h) in bf16 and u in f32. Phase-2 computes
//     c = tanh(Cx[t] + (r*h)@Whc), h = u*h + (1-u)*c. Kernel launch order on
//     the stream provides the grid-wide sync between phases/steps.
//   * Hidden weights pre-packed once into per-lane-contiguous 32x16 WMMA
//     B-tiles (bf16) -> two b128 loads per lane per tile; they total ~20.5MB
//     so the whole recurrent working set lives in the 192MB L2.
//   * WMMA loops use an explicit 2x-unrolled ping-pong pipeline: two
//     independent fragment register sets; loads for k+1 issue into one set
//     while v_wmma consumes the other. Loads write their destination VGPRs
//     directly, so no rotation copies; waits stay partial (loadcnt <= N).
// Workspace requirement: ~266 MB.
// ---------------------------------------------------------------------------

typedef __bf16 bf16;
typedef __attribute__((ext_vector_type(16))) __bf16 v16bf;
typedef __attribute__((ext_vector_type(8)))  __bf16 bf16x8;
typedef __attribute__((ext_vector_type(8)))  float  v8f;

union BFrag { v16bf v16; bf16x8 h[2]; };

// A-fragment: 16x32 bf16 tile of row-major A (lda in elements).
// ISA layout: lanes 0-15 hold row M=lane, K = {0..7, 16..23};
//             lanes 16-31 hold row M=lane-16, K = {8..15, 24..31}.
__device__ __forceinline__ v16bf load_a_tile(const bf16* A, int lda, int mbase,
                                             int kt, int lane) {
  const int r = lane & 15, half = lane >> 4;
  const bf16* p = A + (size_t)(mbase + r) * lda + kt * 32 + half * 8;
  BFrag f;
  f.h[0] = *reinterpret_cast<const bf16x8*>(p);        // K base + {0..7}
  f.h[1] = *reinterpret_cast<const bf16x8*>(p + 16);   // K base + {16..23}
  return f.v16;
}

// B-fragment from pre-packed weights: tile (kt,nt) is 512 bf16, lane's 16
// values contiguous at lane*16 (packed to match dense 32x16 bf16 B layout:
// lane n = lane&15, K = kt*32 + (lane>>4)*8 + (i<8 ? i : i+8)).
__device__ __forceinline__ v16bf load_b_tile(const bf16* Bp, int KT, int nt,
                                             int kt, int lane) {
  const bf16* p = Bp + (((size_t)nt * KT + kt) << 9) + lane * 16;
  BFrag f;
  f.h[0] = *reinterpret_cast<const bf16x8*>(p);
  f.h[1] = *reinterpret_cast<const bf16x8*>(p + 8);
  return f.v16;
}

__device__ __forceinline__ v8f wmma_bf16(v16bf a, v16bf b, v8f c) {
  return __builtin_amdgcn_wmma_f32_16x16x32_bf16(
      /*neg_a=*/false, a, /*neg_b=*/false, b,
      /*c_mod=*/(short)0, c, /*reuse_a=*/false, /*reuse_b=*/false);
}

__device__ __forceinline__ float sigmoidf_fast(float x) {
  return 1.0f / (1.0f + __expf(-x));
}

// ---- one-time prep kernels -------------------------------------------------

// Convert f32 weight [K,N] row-major -> packed bf16 WMMA B-tiles.
__global__ void pack_weight_bf16(const float* __restrict__ W,
                                 bf16* __restrict__ out, int K, int N) {
  int tid  = blockIdx.x * blockDim.x + threadIdx.x;
  int lane = tid & 31;
  int tile = tid >> 5;
  int KT = K >> 5, NT = N >> 4;
  if (tile >= KT * NT) return;
  int kt = tile % KT, nt = tile / KT;
  int n     = nt * 16 + (lane & 15);
  int kbase = kt * 32 + (lane >> 4) * 8;
  bf16* dst = out + ((size_t)tile << 9) + lane * 16;
#pragma unroll
  for (int i = 0; i < 16; ++i)
    dst[i] = (bf16)W[(size_t)(kbase + (i < 8 ? i : i + 8)) * N + n];
}

// Embedding gather, stored s-major: xe[(s*B+b)*E + e] = emb[x[b*S+s]][e].
__global__ void embed_gather(const int* __restrict__ x,
                             const float* __restrict__ emb,
                             bf16* __restrict__ xe, int B, int S, int E) {
  int row = blockIdx.x;           // row = s*B + b
  int s = row / B, b = row % B;
  int tok = x[b * S + s];
  xe[(size_t)row * E + threadIdx.x] =
      (bf16)emb[(size_t)tok * E + threadIdx.x];
}

__global__ void init_h(const float* __restrict__ h0, float* __restrict__ hf,
                       bf16* __restrict__ hb, int n) {
  int i = blockIdx.x * blockDim.x + threadIdx.x;
  if (i < n) { float v = h0[i]; hf[i] = v; hb[i] = (bf16)v; }
}

// ---- big parallel GEMM: C[M,N] = A[M,K](bf16) @ Bpacked + bias -------------
// Block = 256 threads = 8 waves; wave computes a 16(M) x 64(N) strip.
// Ping-pong pipelined over even/odd k-tiles (KT is even: 8 or 32).
__global__ __launch_bounds__(256) void gemm_bias(
    const bf16* __restrict__ A, const bf16* __restrict__ Bp,
    const float* __restrict__ bias, float* __restrict__ C,
    int M, int N, int K) {
  const int lane = threadIdx.x & 31, wave = threadIdx.x >> 5;
  const int mbase = (blockIdx.y * 8 + wave) * 16;
  const int nt0   = blockIdx.x * 4;
  const int KT    = K >> 5;
  v8f acc[4] = {v8f{}, v8f{}, v8f{}, v8f{}};

  v16bf a0 = load_a_tile(A, K, mbase, 0, lane);
  v16bf b0[4];
#pragma unroll
  for (int j = 0; j < 4; ++j) b0[j] = load_b_tile(Bp, KT, nt0 + j, 0, lane);

  for (int kt = 0; kt < KT; kt += 2) {
    if (kt + 6 < KT)  // pull upcoming packed-weight K-tile toward near caches
      __builtin_prefetch(Bp + (((size_t)nt0 * KT + kt + 6) << 9) + lane * 16,
                         0, 3);
    // load odd set (kt+1) while computing even set (kt)
    v16bf a1 = load_a_tile(A, K, mbase, kt + 1, lane);
    v16bf b1[4];
#pragma unroll
    for (int j = 0; j < 4; ++j)
      b1[j] = load_b_tile(Bp, KT, nt0 + j, kt + 1, lane);
#pragma unroll
    for (int j = 0; j < 4; ++j) acc[j] = wmma_bf16(a0, b0[j], acc[j]);
    // load even set (kt+2, clamped on last iter) while computing odd set
    const int kt2 = (kt + 2 < KT) ? kt + 2 : kt;
    a0 = load_a_tile(A, K, mbase, kt2, lane);
#pragma unroll
    for (int j = 0; j < 4; ++j) b0[j] = load_b_tile(Bp, KT, nt0 + j, kt2, lane);
#pragma unroll
    for (int j = 0; j < 4; ++j) acc[j] = wmma_bf16(a1, b1[j], acc[j]);
  }

  const int half = lane >> 4, nl = lane & 15;
#pragma unroll
  for (int j = 0; j < 4; ++j) {
    int ncol = (nt0 + j) * 16 + nl;
    float bv = bias ? bias[ncol] : 0.0f;
#pragma unroll
    for (int v = 0; v < 8; ++v) {
      int row = mbase + v + half * 8;
      C[(size_t)row * N + ncol] = acc[j][v] + bv;
    }
  }
}

// ---- recurrent phase 1: gates ---------------------------------------------
// gi = Gx[t] + h @ Whg ; sigmoid ; r -> store (r*h) bf16, u -> store f32.
// M = 32 (2 tiles), N = 2U = 2048 (128 tiles) -> 256 tiles = 32 blocks.
__global__ __launch_bounds__(256) void gru_gates(
    const bf16* __restrict__ hb, const float* __restrict__ hf,
    const bf16* __restrict__ Wgh, const float* __restrict__ Gx,
    float* __restrict__ ubuf, bf16* __restrict__ rh, int t, int U2, int U) {
  const int lane = threadIdx.x & 31, wave = threadIdx.x >> 5;
  const int tileId = blockIdx.x * 8 + wave;
  const int mbase  = (tileId & 1) * 16;
  const int nTile  = tileId >> 1;
  const int KT     = U >> 5;                        // 32 (even)
  v8f acc{};

  v16bf a0 = load_a_tile(hb, U, mbase, 0, lane);
  v16bf b0 = load_b_tile(Wgh, KT, nTile, 0, lane);
  for (int kt = 0; kt < KT; kt += 2) {
    if (kt + 6 < KT)
      __builtin_prefetch(Wgh + (((size_t)nTile * KT + kt + 6) << 9) + lane * 16,
                         0, 3);
    v16bf a1 = load_a_tile(hb, U, mbase, kt + 1, lane);
    v16bf b1 = load_b_tile(Wgh, KT, nTile, kt + 1, lane);
    acc = wmma_bf16(a0, b0, acc);
    const int kt2 = (kt + 2 < KT) ? kt + 2 : kt;
    a0 = load_a_tile(hb, U, mbase, kt2, lane);
    b0 = load_b_tile(Wgh, KT, nTile, kt2, lane);
    acc = wmma_bf16(a1, b1, acc);
  }

  const int half = lane >> 4;
  const int ncol = nTile * 16 + (lane & 15);
#pragma unroll
  for (int v = 0; v < 8; ++v) {
    int brow = mbase + v + half * 8;                 // batch row 0..31
    float gi = acc[v] + Gx[(size_t)(t * 32 + brow) * U2 + ncol];
    float g  = sigmoidf_fast(gi);
    if (ncol < U) {                                  // r gate -> r*h (bf16)
      size_t hi = (size_t)brow * U + ncol;
      rh[hi] = (bf16)(g * hf[hi]);
    } else {                                         // u gate -> f32
      ubuf[(size_t)brow * U + (ncol - U)] = g;
    }
  }
}

// ---- recurrent phase 2: candidate + state update --------------------------
// c = tanh(Cx[t] + (r*h) @ Whc) ; h = u*h + (1-u)*c ; emit y.
// M = 32 (2 tiles), N = U = 1024 (64 tiles) -> 128 tiles = 16 blocks.
__global__ __launch_bounds__(256) void gru_update(
    const bf16* __restrict__ rh, const bf16* __restrict__ Wch,
    const float* __restrict__ Cx, const float* __restrict__ ubuf,
    float* __restrict__ hf, bf16* __restrict__ hb,
    bf16* __restrict__ y_bf16, float* __restrict__ y_f32,
    int t, int S, int U) {
  const int lane = threadIdx.x & 31, wave = threadIdx.x >> 5;
  const int tileId = blockIdx.x * 8 + wave;
  const int mbase  = (tileId & 1) * 16;
  const int nTile  = tileId >> 1;
  const int KT     = U >> 5;                        // 32 (even)
  v8f acc{};

  v16bf a0 = load_a_tile(rh, U, mbase, 0, lane);
  v16bf b0 = load_b_tile(Wch, KT, nTile, 0, lane);
  for (int kt = 0; kt < KT; kt += 2) {
    if (kt + 6 < KT)
      __builtin_prefetch(Wch + (((size_t)nTile * KT + kt + 6) << 9) + lane * 16,
                         0, 3);
    v16bf a1 = load_a_tile(rh, U, mbase, kt + 1, lane);
    v16bf b1 = load_b_tile(Wch, KT, nTile, kt + 1, lane);
    acc = wmma_bf16(a0, b0, acc);
    const int kt2 = (kt + 2 < KT) ? kt + 2 : kt;
    a0 = load_a_tile(rh, U, mbase, kt2, lane);
    b0 = load_b_tile(Wch, KT, nTile, kt2, lane);
    acc = wmma_bf16(a1, b1, acc);
  }

  const int half = lane >> 4;
  const int ncol = nTile * 16 + (lane & 15);
#pragma unroll
  for (int v = 0; v < 8; ++v) {
    int brow = mbase + v + half * 8;                 // batch row 0..31
    float c  = tanhf(acc[v] + Cx[(size_t)(t * 32 + brow) * U + ncol]);
    size_t hi = (size_t)brow * U + ncol;
    float u    = ubuf[hi];
    float hnew = u * hf[hi] + (1.0f - u) * c;        // TF GRU update
    hf[hi] = hnew;
    hb[hi] = (bf16)hnew;
    if (y_bf16) y_bf16[(size_t)(t * 32 + brow) * U + ncol] = (bf16)hnew;
    if (y_f32)  y_f32[((size_t)brow * S + t) * U + ncol]   = hnew;
  }
}

// ---------------------------------------------------------------------------

extern "C" void kernel_launch(void* const* d_in, const int* in_sizes, int n_in,
                              void* d_out, int out_size, void* d_ws,
                              size_t ws_size, hipStream_t stream) {
  (void)in_sizes; (void)n_in; (void)out_size; (void)ws_size;
  constexpr int B = 32, S = 512, E = 256, U = 1024;

  const int*   x    = (const int*)d_in[0];
  const float* h0_0 = (const float*)d_in[1];
  const float* h0_1 = (const float*)d_in[2];
  const float* emb  = (const float*)d_in[3];
  const float* Wg0  = (const float*)d_in[4];
  const float* bg0  = (const float*)d_in[5];
  const float* Wc0  = (const float*)d_in[6];
  const float* bc0  = (const float*)d_in[7];
  const float* Wg1  = (const float*)d_in[8];
  const float* bg1  = (const float*)d_in[9];
  const float* Wc1  = (const float*)d_in[10];
  const float* bc1  = (const float*)d_in[11];

  float* y1_out = (float*)d_out;                     // [B,S,U]
  float* s0_out = y1_out + (size_t)B * S * U;        // [B,U]
  float* s1_out = s0_out + (size_t)B * U;            // [B,U]

  char* wsp = (char*)d_ws;
  size_t off = 0;
  auto carve = [&](size_t bytes) -> char* {
    char* p = wsp + off;
    off += (bytes + 255) & ~(size_t)255;
    return p;
  };
  bf16*  xe    = (bf16*)carve((size_t)S * B * E * 2);       //   8 MB
  bf16*  pWg0x = (bf16*)carve((size_t)E * 2 * U * 2);
  bf16*  pWg0h = (bf16*)carve((size_t)U * 2 * U * 2);
  bf16*  pWc0x = (bf16*)carve((size_t)E * U * 2);
  bf16*  pWc0h = (bf16*)carve((size_t)U * U * 2);
  bf16*  pWg1x = (bf16*)carve((size_t)U * 2 * U * 2);
  bf16*  pWg1h = (bf16*)carve((size_t)U * 2 * U * 2);
  bf16*  pWc1x = (bf16*)carve((size_t)U * U * 2);
  bf16*  pWc1h = (bf16*)carve((size_t)U * U * 2);           // ~20.5 MB packed
  float* Gx    = (float*)carve((size_t)S * B * 2 * U * 4);  // 134 MB
  float* Cx    = (float*)carve((size_t)S * B * U * 4);      //  67 MB
  bf16*  y0    = (bf16*)carve((size_t)S * B * U * 2);       //  33.5 MB
  float* hf    = (float*)carve((size_t)B * U * 4);
  bf16*  hb    = (bf16*)carve((size_t)B * U * 2);
  bf16*  rh    = (bf16*)carve((size_t)B * U * 2);
  float* ubuf  = (float*)carve((size_t)B * U * 4);

  auto pack = [&](const float* src, bf16* dst, int K, int N) {
    int threads = (K / 32) * (N / 16) * 32;
    pack_weight_bf16<<<(threads + 255) / 256, 256, 0, stream>>>(src, dst, K, N);
  };
  // W = [Wx ; Wh] row-blocks are contiguous in the source matrices.
  pack(Wg0,                      pWg0x, E, 2 * U);
  pack(Wg0 + (size_t)E * 2 * U,  pWg0h, U, 2 * U);
  pack(Wc0,                      pWc0x, E, U);
  pack(Wc0 + (size_t)E * U,      pWc0h, U, U);
  pack(Wg1,                      pWg1x, U, 2 * U);
  pack(Wg1 + (size_t)U * 2 * U,  pWg1h, U, 2 * U);
  pack(Wc1,                      pWc1x, U, U);
  pack(Wc1 + (size_t)U * U,      pWc1h, U, U);

  embed_gather<<<S * B, E, 0, stream>>>(x, emb, xe, B, S, E);

  // ---- layer 0 ----
  gemm_bias<<<dim3(2 * U / 64, S * B / 128), 256, 0, stream>>>(
      xe, pWg0x, bg0, Gx, S * B, 2 * U, E);
  gemm_bias<<<dim3(U / 64, S * B / 128), 256, 0, stream>>>(
      xe, pWc0x, bc0, Cx, S * B, U, E);
  init_h<<<(B * U + 255) / 256, 256, 0, stream>>>(h0_0, hf, hb, B * U);
  for (int t = 0; t < S; ++t) {
    gru_gates<<<32, 256, 0, stream>>>(hb, hf, pWg0h, Gx, ubuf, rh, t, 2 * U, U);
    gru_update<<<16, 256, 0, stream>>>(rh, pWc0h, Cx, ubuf, hf, hb,
                                       y0, nullptr, t, S, U);
  }
  hipMemcpyAsync(s0_out, hf, (size_t)B * U * 4, hipMemcpyDeviceToDevice,
                 stream);

  // ---- layer 1 ----
  gemm_bias<<<dim3(2 * U / 64, S * B / 128), 256, 0, stream>>>(
      y0, pWg1x, bg1, Gx, S * B, 2 * U, U);
  gemm_bias<<<dim3(U / 64, S * B / 128), 256, 0, stream>>>(
      y0, pWc1x, bc1, Cx, S * B, U, U);
  init_h<<<(B * U + 255) / 256, 256, 0, stream>>>(h0_1, hf, hb, B * U);
  for (int t = 0; t < S; ++t) {
    gru_gates<<<32, 256, 0, stream>>>(hb, hf, pWg1h, Gx, ubuf, rh, t, 2 * U, U);
    gru_update<<<16, 256, 0, stream>>>(rh, pWc1h, Cx, ubuf, hf, hb,
                                       nullptr, y1_out, t, S, U);
  }
  hipMemcpyAsync(s1_out, hf, (size_t)B * U * 4, hipMemcpyDeviceToDevice,
                 stream);
}
